// MultiHeadGATLayer_64690797412824
// MI455X (gfx1250) — matrix operand order repaired and checked by reading, not compiled, not run on hardware
//
#include <hip/hip_runtime.h>
#include <hip/hip_bf16.h>

static constexpr int kN    = 6144;
static constexpr int kFIN  = 128;
static constexpr int kFOUT = 64;
static constexpr int kH    = 2;
static constexpr float kALPHA = 0.2f;
static constexpr float kEPS   = 1e-5f;
static constexpr float kNEG   = -9000000000000000.0f;

typedef float  v2f   __attribute__((ext_vector_type(2)));
typedef float  v4f   __attribute__((ext_vector_type(4)));
typedef float  v8f   __attribute__((ext_vector_type(8)));
typedef int    v4i   __attribute__((ext_vector_type(4)));
typedef __bf16 v8bf  __attribute__((ext_vector_type(8)));
typedef __bf16 v16bf __attribute__((ext_vector_type(16)));

#ifndef __has_builtin
#define __has_builtin(x) 0
#endif

#if __has_builtin(__builtin_amdgcn_global_load_async_to_lds_b128)
#define ASYNC_LDS 1
// The builtin's parameters are typed, address-space-qualified vector pointers:
// (global v4i* src, local v4i* dst, constant int offset, constant int cpol)
typedef __attribute__((address_space(1))) v4i* g_v4i_ptr;
typedef __attribute__((address_space(3))) v4i* l_v4i_ptr;
#else
#define ASYNC_LDS 0
#endif

// 16-byte global -> LDS copy, via the async DMA path when available.
__device__ __forceinline__ void asyncCopy16(void* lds, const void* g) {
#if ASYNC_LDS
  __builtin_amdgcn_global_load_async_to_lds_b128((g_v4i_ptr)g, (l_v4i_ptr)lds,
                                                 0, 0);
#else
  *(v4i*)lds = *(const v4i*)g;
#endif
}

__device__ __forceinline__ void asyncWaitLE1() {
#if ASYNC_LDS
  asm volatile("s_wait_asynccnt 0x1" ::: "memory");
#endif
}
__device__ __forceinline__ void asyncWait0() {
#if ASYNC_LDS
  asm volatile("s_wait_asynccnt 0x0" ::: "memory");
#endif
}

// ---------------------------------------------------------------------------
// K1: ht[h] = h @ W[h]   (N x FIN)@(FIN x FOUT), f32 WMMA 16x16x4.
// One wave per 16x16 output tile; 3072 tiles; 4 waves / block.
// ---------------------------------------------------------------------------
__global__ __launch_bounds__(128)
void ht_gemm_kernel(const float* __restrict__ hmat,
                    const float* __restrict__ W,
                    float* __restrict__ ht) {
  const int lane = threadIdx.x & 31;
  const int wave = threadIdx.x >> 5;
  const int tile = blockIdx.x * 4 + wave;          // 0..3071
  const int co   = tile & 3;                       // FOUT/16 = 4
  const int rt   = (tile >> 2) % (kN / 16);        // N/16 = 384
  const int hd   = tile / (4 * (kN / 16));         // head
  const int rowBase = rt * 16;
  const int colBase = co * 16;
  const int half = lane >> 4;                      // 0: K pair {0,1}, 1: {2,3}
  const int l15  = lane & 15;
  const float* Wh = W + hd * kFIN * kFOUT;

  v8f acc = {};
  for (int k0 = 0; k0 < kFIN; k0 += 4) {
    const int ka = k0 + 2 * half;
    // A 16x4 f32: lanes 0-15 hold K=0,1; lanes 16-31 hold K=2,3 (M = lane%16)
    v2f a = *(const v2f*)(hmat + (rowBase + l15) * kFIN + ka);
    // B 4x16 f32: K rows striped 2-per-VGPR, lanes 0-15 = K{0,1}, 16-31 = K{2,3}
    v2f b;
    b.x = Wh[ka * kFOUT + colBase + l15];
    b.y = Wh[(ka + 1) * kFOUT + colBase + l15];
    acc = __builtin_amdgcn_wmma_f32_16x16x4_f32(false, a, false, b,
                                                (short)0, acc, false, false);
  }
  for (int r = 0; r < 8; ++r) {
    const int row = rowBase + r + half * 8;
    ht[(hd * kN + row) * kFOUT + colBase + l15] = acc[r];
  }
}

// ---------------------------------------------------------------------------
// K1b: rewrite ht into bf16 B-fragment-major order for the big GEMM.
// Fragment = 32x16 (KxN) tile; element (lane, j): k = 16*(lane>>4)+j,
// n = lane&15; fragments indexed ((hd*(N/32)+kb)*4 + colTile).
// After this, K4 fetches each lane's B operand as two contiguous v8bf loads.
// ---------------------------------------------------------------------------
__global__ __launch_bounds__(256)
void swizzleB_kernel(const float* __restrict__ ht, __bf16* __restrict__ wsB) {
  const int t = blockIdx.x * 256 + threadIdx.x;    // < H*N*FOUT = 786432
  const int j    = t & 15;
  const int lane = (t >> 4) & 31;
  const int ct   = (t >> 9) & 3;
  const int kb   = (t >> 11) % (kN / 32);
  const int hd   = t / (2048 * (kN / 32));
  const int k = kb * 32 + (lane >> 4) * 16 + j;
  const int n = ct * 16 + (lane & 15);
  wsB[t] = (__bf16)ht[((size_t)hd * kN + k) * kFOUT + n];
}

// ---------------------------------------------------------------------------
// K2: s1[h,n] = ht[h,n,:] . a1[h]   s2 likewise. One thread per (h,n).
// ---------------------------------------------------------------------------
__global__ void attn_score_kernel(const float* __restrict__ ht,
                                  const float* __restrict__ a,
                                  float* __restrict__ s1,
                                  float* __restrict__ s2) {
  const int idx = blockIdx.x * blockDim.x + threadIdx.x;
  if (idx >= kH * kN) return;
  const int hd = idx / kN;
  const float* row = ht + (size_t)idx * kFOUT;
  const float* a1  = a + hd * 2 * kFOUT;
  const float* a2  = a1 + kFOUT;
  float t1 = 0.f, t2 = 0.f;
  for (int o = 0; o < kFOUT; ++o) {
    const float v = row[o];
    t1 += v * a1[o];
    t2 += v * a2[o];
  }
  s1[idx] = t1;
  s2[idx] = t2;
}

// ---------------------------------------------------------------------------
// K3: per node n, BOTH heads: e = leaky(s1+s2) masked by adj, softmax over m,
// att rows -> d_out. adj row staged through double-buffered async-to-LDS
// chunks (ASYNCcnt-tracked DMA), read from HBM exactly once.
// ---------------------------------------------------------------------------
__global__ __launch_bounds__(256)
void softmax_kernel(const int* __restrict__ adj,
                    const float* __restrict__ s1,
                    const float* __restrict__ s2,
                    float* __restrict__ att) {
  __shared__ float eL0[kN];
  __shared__ float eL1[kN];
  __shared__ float r0[256];
  __shared__ float r1[256];
  __shared__ __align__(16) int abuf[2][1024];      // 2 x 4KB chunks
  constexpr int CH = 1024, NCH = kN / CH;          // 6 chunks

  const int n = blockIdx.x;
  const int t = threadIdx.x;
  const int* adjRow = adj + (size_t)n * kN;
  const float s1h0 = s1[n];
  const float s1h1 = s1[kN + n];

  // prime chunk 0: each thread DMAs 16 bytes (4 ints)
  asyncCopy16(&abuf[0][t * 4], adjRow + t * 4);

  float mx0 = kNEG, mx1 = kNEG;
  for (int c = 0; c < NCH; ++c) {
    if (c + 1 < NCH) {
      asyncCopy16(&abuf[(c + 1) & 1][t * 4], adjRow + (c + 1) * CH + t * 4);
      asyncWaitLE1();                              // chunk c complete (in-order)
    } else {
      asyncWait0();
    }
    __syncthreads();                               // all waves' DMA visible
    const int* ab = abuf[c & 1];
    for (int i = 0; i < 4; ++i) {
      const int m  = c * CH + t * 4 + i;
      const int av = ab[t * 4 + i];
      float v0 = s1h0 + s2[m];
      v0 = v0 > 0.f ? v0 : kALPHA * v0;
      float v1 = s1h1 + s2[kN + m];
      v1 = v1 > 0.f ? v1 : kALPHA * v1;
      const float e0 = av > 0 ? v0 : kNEG;
      const float e1 = av > 0 ? v1 : kNEG;
      eL0[m] = e0;
      eL1[m] = e1;
      mx0 = fmaxf(mx0, e0);
      mx1 = fmaxf(mx1, e1);
    }
    __syncthreads();                               // done reading before reuse
  }

  r0[t] = mx0; r1[t] = mx1;
  __syncthreads();
  for (int s = 128; s > 0; s >>= 1) {
    if (t < s) { r0[t] = fmaxf(r0[t], r0[t + s]); r1[t] = fmaxf(r1[t], r1[t + s]); }
    __syncthreads();
  }
  const float m0 = r0[0], m1 = r1[0];
  __syncthreads();

  float sum0 = 0.f, sum1 = 0.f;
  for (int m = t; m < kN; m += 256) {
    sum0 += __expf(eL0[m] - m0);
    sum1 += __expf(eL1[m] - m1);
  }
  r0[t] = sum0; r1[t] = sum1;
  __syncthreads();
  for (int s = 128; s > 0; s >>= 1) {
    if (t < s) { r0[t] += r0[t + s]; r1[t] += r1[t + s]; }
    __syncthreads();
  }
  const float inv0 = 1.f / r0[0];
  const float inv1 = 1.f / r1[0];

  float* a0 = att + (size_t)n * kN;
  float* a1 = att + (size_t)kN * kN + (size_t)n * kN;
  for (int m = t; m < kN; m += 256) {
    a0[m] = __expf(eL0[m] - m0) * inv0;
    a1[m] = __expf(eL1[m] - m1) * inv1;
  }
}

// ---------------------------------------------------------------------------
// K4: hp[h] = att[h] @ ht[h]   (N x N)@(N x 64), bf16 WMMA 16x16x32 / f32 acc.
// Block = 4 waves, 16 rows x 64 cols.  att tile: 2x global_load_b128 +
// convert + 1x ds_store_b128 per thread; A fragment: 2x ds_load_b128;
// B fragment: 2x coalesced v8bf loads from the swizzled buffer.
// ---------------------------------------------------------------------------
__global__ __launch_bounds__(128)
void hp_gemm_kernel(const float* __restrict__ att,
                    const __bf16* __restrict__ wsB,
                    float* __restrict__ hp) {
  __shared__ __align__(16) __bf16 atile[16 * 72];  // stride 72 elems = 144B
  const int hd = blockIdx.x / (kN / 16);
  const int rt = blockIdx.x % (kN / 16);
  const int rowBase = rt * 16;
  const int lane = threadIdx.x & 31;
  const int wave = threadIdx.x >> 5;
  const int colBase = wave * 16;
  const int half = lane >> 4;
  const int l15  = lane & 15;
  const int trow = threadIdx.x >> 3;               // 0..15
  const int tcol = (threadIdx.x & 7) * 8;          // 0,8,...,56
  const float* attH = att + (size_t)hd * kN * kN;

  v8f acc = {};
  for (int k0 = 0; k0 < kN; k0 += 64) {
    // stage 16x64 att tile, f32 -> bf16
    const float* src = attH + (size_t)(rowBase + trow) * kN + k0 + tcol;
    v4f f0 = *(const v4f*)(src);
    v4f f1 = *(const v4f*)(src + 4);
    v8bf pk;
    pk[0] = (__bf16)f0[0]; pk[1] = (__bf16)f0[1];
    pk[2] = (__bf16)f0[2]; pk[3] = (__bf16)f0[3];
    pk[4] = (__bf16)f1[0]; pk[5] = (__bf16)f1[1];
    pk[6] = (__bf16)f1[2]; pk[7] = (__bf16)f1[3];
    *(v8bf*)(atile + trow * 72 + tcol) = pk;
    __syncthreads();

    for (int kk = 0; kk < 64; kk += 32) {
      // A (16-bit 16x32): lane holds 16 contiguous K (k = 16*half + s).
      const __bf16* ar = atile + l15 * 72 + kk;
      v8bf alo = *(const v8bf*)(ar + 16 * half);
      v8bf ahi = *(const v8bf*)(ar + 16 * half + 8);
      v16bf a = __builtin_shufflevector(alo, ahi, 0, 1, 2, 3, 4, 5, 6, 7,
                                        8, 9, 10, 11, 12, 13, 14, 15);
      // B (16-bit 32x16): fragment-major buffer, 16 contiguous bf16 per lane.
      const int kb = (k0 + kk) >> 5;
      const __bf16* bp =
          wsB + ((size_t)((hd * (kN / 32) + kb) * 4 + wave) << 9) + lane * 16;
      v8bf blo = *(const v8bf*)(bp);
      v8bf bhi = *(const v8bf*)(bp + 8);
      v16bf b = __builtin_shufflevector(blo, bhi, 0, 1, 2, 3, 4, 5, 6, 7,
                                        8, 9, 10, 11, 12, 13, 14, 15);
      acc = __builtin_amdgcn_wmma_f32_16x16x32_bf16(false, a, false, b,
                                                    (short)0, acc, false, false);
    }
    __syncthreads();
  }
  for (int r = 0; r < 8; ++r) {
    const int row = rowBase + r + half * 8;
    hp[((size_t)hd * kN + row) * kFOUT + colBase + l15] = acc[r];
  }
}

// ---------------------------------------------------------------------------
// K5: out[n, h*64+o] = LN(hp[h,n,o]) * gamma + beta.  One 128-thr block / node.
// ---------------------------------------------------------------------------
__global__ __launch_bounds__(128)
void layernorm_kernel(const float* __restrict__ hp,
                      const float* __restrict__ gamma,
                      const float* __restrict__ beta,
                      float* __restrict__ out) {
  __shared__ float red[128];
  const int n = blockIdx.x;
  const int f = threadIdx.x;
  const int hd = f >> 6, o = f & 63;
  const float v = hp[((size_t)hd * kN + n) * kFOUT + o];
  red[f] = v;
  __syncthreads();
  for (int s = 64; s > 0; s >>= 1) { if (f < s) red[f] += red[f + s]; __syncthreads(); }
  const float mu = red[0] * (1.f / 128.f);
  __syncthreads();
  const float d = v - mu;
  red[f] = d * d;
  __syncthreads();
  for (int s = 64; s > 0; s >>= 1) { if (f < s) red[f] += red[f + s]; __syncthreads(); }
  const float var = red[0] * (1.f / 128.f);
  out[(size_t)n * 128 + f] = d * rsqrtf(var + kEPS) * gamma[f] + beta[f];
}

// ---------------------------------------------------------------------------
extern "C" void kernel_launch(void* const* d_in, const int* in_sizes, int n_in,
                              void* d_out, int out_size, void* d_ws, size_t ws_size,
                              hipStream_t stream) {
  const float* h     = (const float*)d_in[0];
  const int*   adj   = (const int*)d_in[1];
  const float* W     = (const float*)d_in[2];
  const float* a     = (const float*)d_in[3];
  const float* gamma = (const float*)d_in[4];
  const float* beta  = (const float*)d_in[5];

  float* out = (float*)d_out;                         // (N, H*FOUT)
  float* att = out + (size_t)kN * kH * kFOUT;         // (H, N, N)

  float*  ht  = (float*)d_ws;                         // H*N*FOUT f32
  float*  hp  = ht + (size_t)kH * kN * kFOUT;         // H*N*FOUT f32
  float*  s1  = hp + (size_t)kH * kN * kFOUT;         // H*N
  float*  s2  = s1 + (size_t)kH * kN;                 // H*N
  __bf16* wsB = (__bf16*)(s2 + (size_t)kH * kN);      // H*N*FOUT bf16 (swizzled)

  ht_gemm_kernel<<<dim3(kH * (kN / 16) * (kFOUT / 16) / 4), dim3(128), 0, stream>>>(
      h, W, ht);
  swizzleB_kernel<<<dim3(kH * kN * kFOUT / 256), dim3(256), 0, stream>>>(ht, wsB);
  attn_score_kernel<<<dim3((kH * kN + 255) / 256), dim3(256), 0, stream>>>(
      ht, a, s1, s2);
  softmax_kernel<<<dim3(kN), dim3(256), 0, stream>>>(adj, s1, s2, att);
  hp_gemm_kernel<<<dim3(kH * (kN / 16)), dim3(128), 0, stream>>>(att, wsB, hp);
  layernorm_kernel<<<dim3(kN), dim3(128), 0, stream>>>(hp, gamma, beta, out);
}
